// ftat_BiLSTM_44384192037502
// MI455X (gfx1250) — matrix operand order repaired and checked by reading, not compiled, not run on hardware
//
#include <hip/hip_runtime.h>

// ---- problem dims ----
constexpr int NB = 2048;   // batch
constexpr int NS = 96;     // seq
constexpr int ND = 7;      // features
// H = 64, H2 = 16 hardcoded below

typedef __attribute__((ext_vector_type(16))) __bf16         v16bf;
typedef __attribute__((ext_vector_type(8)))  float          v8f;
typedef __attribute__((ext_vector_type(16))) unsigned short v16u;
typedef __attribute__((ext_vector_type(8)))  unsigned short v8u;
typedef __attribute__((ext_vector_type(4)))  unsigned int   v4ui;
typedef __attribute__((ext_vector_type(8)))  int            v8i;
typedef __attribute__((ext_vector_type(4)))  int            v4i;

__device__ __forceinline__ unsigned short f2bf(float f) {
  union { float f; unsigned int u; } v; v.f = f;
  unsigned int u = v.u + 0x7FFFu + ((v.u >> 16) & 1u);   // round-to-nearest-even
  return (unsigned short)(u >> 16);
}
__device__ __forceinline__ float bf2f(unsigned short h) {
  union { unsigned int u; float f; } v; v.u = ((unsigned int)h) << 16;
  return v.f;
}
__device__ __forceinline__ float sigmoidf(float x) { return 1.0f / (1.0f + __expf(-x)); }

// ---- WMMA fragment loaders (CDNA5 wave32 layouts, CDNA5 ISA) ----
// 16-bit A 16x32: lanes 0-15 rows 0-15; elem e -> K = 16*half + e (file's own k order, not ISA table).
//                 Two contiguous 16B runs.
__device__ __forceinline__ v16bf load_A_lds(const unsigned short* base, int ldk) {
  int lane = threadIdx.x & 31;
  int m = lane & 15;
  int h8 = (lane & 16) >> 1;                 // 0 or 8
  const unsigned short* row = base + m * ldk + 2 * h8;
  v8u lo = *(const v8u*)(row);
  v8u hi = *(const v8u*)(row + 8);
  v16u u;
#pragma unroll
  for (int e = 0; e < 8; ++e) { u[e] = lo[e]; u[e + 8] = hi[e]; }
  return __builtin_bit_cast(v16bf, u);
}
// B 32x16 pre-swizzled blob: each lane reads its 16 bf16 (32B) contiguously.
__device__ __forceinline__ v16bf load_B_swz(const unsigned short* blob) {
  int lane = threadIdx.x & 31;
  return __builtin_bit_cast(v16bf, *(const v16u*)(blob + lane * 16));
}
// B 32x16 from natural [K][N] storage: lanes 0-15 K=0..15, 16-31 K=16..31, N=lane&15.
__device__ __forceinline__ v16bf load_B_nat(const unsigned short* base, int ldn) {
  int lane = threadIdx.x & 31;
  int n = lane & 15;
  int kb = (lane & 16);                      // 0 or 16
  v16u u;
#pragma unroll
  for (int e = 0; e < 16; ++e) u[e] = base[(kb + e) * ldn + n];
  return __builtin_bit_cast(v16bf, u);
}

#define WMMA_BF16(a, b, c) \
  __builtin_amdgcn_wmma_f32_16x16x32_bf16(false, (a), false, (b), (short)0, (c), false, false)

// =====================================================================
// Kernel 0: weight prep — fold biases, bf16-convert, pre-swizzle into
// per-(nt,kc) B-fragment blobs of 512 elems (32 lanes x 16).
// =====================================================================
__global__ void prep_kernel(const float* l1f_wih, const float* l1f_whh,
                            const float* l1f_bih, const float* l1f_bhh,
                            const float* l1b_wih, const float* l1b_whh,
                            const float* l1b_bih, const float* l1b_bhh,
                            const float* tv_w, const float* tk_w, const float* tq_w,
                            const float* l2_wih, const float* l2_whh,
                            const float* l2_bih, const float* l2_bhh,
                            unsigned short* wc1, float* b1,
                            unsigned short* wc2, float* b2,
                            unsigned short* wt) {
  int tid = blockIdx.x * blockDim.x + threadIdx.x;
  int nth = gridDim.x * blockDim.x;
  // LSTM1 combined weights: K = [x(7) | h(64) | pad] = 96, N = 256, 2 dirs
  for (int i = tid; i < 2 * 16 * 3 * 512; i += nth) {
    int d = i / 24576, r = i % 24576;
    int f = r >> 9, nt = f / 3, kc = f % 3;
    int lane = (r >> 4) & 31, e = r & 15;
    int n = nt * 16 + (lane & 15);
    int k = kc * 32 + ((lane >> 4) << 4) + e;
    const float* wih = d ? l1b_wih : l1f_wih;
    const float* whh = d ? l1b_whh : l1f_whh;
    float val = (k < 7) ? wih[n * 7 + k] : ((k < 71) ? whh[n * 64 + (k - 7)] : 0.0f);
    wc1[i] = f2bf(val);
  }
  // LSTM2 combined weights: K = [to(64) | h(16) | pad] = 96, N = 64
  for (int i = tid; i < 4 * 3 * 512; i += nth) {
    int f = i >> 9, nt = f / 3, kc = f % 3;
    int lane = (i >> 4) & 31, e = i & 15;
    int n = nt * 16 + (lane & 15);
    int k = kc * 32 + ((lane >> 4) << 4) + e;
    float val = (k < 64) ? l2_wih[n * 64 + k] : ((k < 80) ? l2_whh[n * 16 + (k - 64)] : 0.0f);
    wc2[i] = f2bf(val);
  }
  // temporal projection weights (B[k][n] = w[n][k]): K = 64, N = 64, 3 projections
  for (int i = tid; i < 3 * 4 * 2 * 512; i += nth) {
    int p = i / 4096, r = i % 4096;
    int f = r >> 9, nt = f >> 1, kc = f & 1;
    int lane = (r >> 4) & 31, e = r & 15;
    int n = nt * 16 + (lane & 15);
    int k = kc * 32 + ((lane >> 4) << 4) + e;
    const float* wp = (p == 0) ? tv_w : ((p == 1) ? tk_w : tq_w);
    wt[i] = f2bf(wp[n * 64 + k]);
  }
  for (int i = tid; i < 512; i += nth) {
    int d = i >> 8, n = i & 255;
    b1[i] = d ? (l1b_bih[n] + l1b_bhh[n]) : (l1f_bih[n] + l1f_bhh[n]);
  }
  for (int i = tid; i < 64; i += nth) b2[i] = l2_bih[i] + l2_bhh[i];
}

// =====================================================================
// Kernel 1: feature attention (D=7 — VALU; too small for WMMA tiles).
// One block per batch row. Emits fa as bf16, stride 8 (padded).
// =====================================================================
__global__ void __launch_bounds__(128) featatt_kernel(
    const float* x, const float* fvw, const float* fvb, const float* fkw,
    const float* fkb, const float* fqw, const float* fqb, unsigned short* fa_g) {
  __shared__ float xs[NS][8], vs[NS][8], ks[NS][8], qs[NS][8];
  __shared__ float fm[ND][8];
  int b = blockIdx.x, t = threadIdx.x;
  const float* xb = x + (size_t)b * NS * ND;
  for (int i = t; i < NS * ND; i += 128) xs[i / 7][i % 7] = xb[i];
  __syncthreads();
  if (t < NS) {
    float xv[7], vv[7], kv[7], tmp[7];
    for (int d = 0; d < 7; ++d) xv[d] = xs[t][d];
    for (int i = 0; i < 7; ++i) { float s = fvb[i]; for (int d = 0; d < 7; ++d) s += fvw[i*7+d]*xv[d]; vv[i]=s; vs[t][i]=s; }
    for (int d = 0; d < 7; ++d) tmp[d] = xv[d] + vv[d];
    for (int i = 0; i < 7; ++i) { float s = fkb[i]; for (int d = 0; d < 7; ++d) s += fkw[i*7+d]*tmp[d]; kv[i]=s; ks[t][i]=s; }
    for (int d = 0; d < 7; ++d) tmp[d] = xv[d] + kv[d];
    for (int i = 0; i < 7; ++i) { float s = fqb[i]; for (int d = 0; d < 7; ++d) s += fqw[i*7+d]*tmp[d]; qs[t][i]=s; }
  }
  __syncthreads();
  if (t < 49) {
    int i = t / 7, j = t % 7;
    float s = 0.f;
    for (int ss = 0; ss < NS; ++ss) s += qs[ss][i] * ks[ss][j];
    fm[i][j] = s;
  }
  __syncthreads();
  if (t < 7) {
    float mx = fm[t][0];
    for (int j = 1; j < 7; ++j) mx = fmaxf(mx, fm[t][j]);
    float sum = 0.f, e[7];
    for (int j = 0; j < 7; ++j) { e[j] = __expf(fm[t][j] - mx); sum += e[j]; }
    float inv = 1.0f / sum;
    for (int j = 0; j < 7; ++j) fm[t][j] = e[j] * inv;
  }
  __syncthreads();
  if (t < NS) {
    unsigned short* o = fa_g + ((size_t)b * NS + t) * 8;
    for (int j = 0; j < 7; ++j) {
      float s = xs[t][j];
      for (int i = 0; i < 7; ++i) s += vs[t][i] * fm[i][j];
      o[j] = f2bf(tanhf(s));
    }
    o[7] = 0;
  }
}

// =====================================================================
// Kernel 2: BiLSTM. grid (NB/16, 2 dirs), 256 thr (8 waves). A-tile
// [16 x 96] = [fa | h | 0] in LDS; gates via 48 bf16 WMMAs per step.
// =====================================================================
__global__ void __launch_bounds__(256) lstm1_kernel(
    const unsigned short* fa_g, const unsigned short* wc1, const float* b1,
    unsigned short* hf_g, unsigned short* hb_g) {
  __shared__ __align__(16) unsigned short sA[16 * 96];
  __shared__ float sG[16 * 256];
  __shared__ float sC[16 * 64];
  __shared__ float sB[256];
  int t = threadIdx.x, wv = t >> 5, lane = t & 31;
  int m = lane & 15, h8 = (lane & 16) >> 1;
  int b0 = blockIdx.x * 16;
  int dir = blockIdx.y;
  const unsigned short* wc = wc1 + dir * 24576;
  unsigned short* hout = dir ? hb_g : hf_g;
  __builtin_prefetch((const char*)wc + t * 192, 0, 3);   // warm 49KB weight blob in L2
  for (int i = t; i < 16 * 96; i += 256) sA[i] = 0;
  for (int i = t; i < 16 * 64; i += 256) sC[i] = 0.f;
  if (t < 256) sB[t] = b1[dir * 256 + t];
  __syncthreads();
  for (int st = 0; st < NS; ++st) {
    int s = dir ? (NS - 1 - st) : st;
    if (t < 16) {  // stage x features into A-tile cols 0..6
      v8u r = *(const v8u*)(fa_g + ((size_t)(b0 + t) * NS + s) * 8);
#pragma unroll
      for (int c = 0; c < 7; ++c) sA[t * 96 + c] = r[c];
    }
    __syncthreads();
    v8f acc0 = {}, acc1 = {};
    int nt0 = wv * 2, nt1 = nt0 + 1;
#pragma unroll
    for (int kc = 0; kc < 3; ++kc) {
      v16bf A  = load_A_lds(sA + kc * 32, 96);
      v16bf B0 = load_B_swz(wc + (nt0 * 3 + kc) * 512);
      acc0 = WMMA_BF16(A, B0, acc0);
      v16bf B1 = load_B_swz(wc + (nt1 * 3 + kc) * 512);
      acc1 = WMMA_BF16(A, B1, acc1);
    }
#pragma unroll
    for (int e = 0; e < 8; ++e) {
      int row = e + h8;
      sG[row * 256 + nt0 * 16 + m] = acc0[e];
      sG[row * 256 + nt1 * 16 + m] = acc1[e];
    }
    __syncthreads();
    for (int u = t; u < 16 * 64; u += 256) {   // i,f,g,o gate order
      int r = u >> 6, j = u & 63;
      float gi = sG[r * 256 + j]       + sB[j];
      float gf = sG[r * 256 + 64 + j]  + sB[64 + j];
      float gg = sG[r * 256 + 128 + j] + sB[128 + j];
      float go = sG[r * 256 + 192 + j] + sB[192 + j];
      float c = sigmoidf(gf) * sC[u] + sigmoidf(gi) * tanhf(gg);
      sC[u] = c;
      unsigned short h16 = f2bf(sigmoidf(go) * tanhf(c));
      sA[r * 96 + 7 + j] = h16;   // feed h back into A-tile (bf16)
      hout[((size_t)(b0 + r) * NS + s) * 64 + j] = h16;
    }
    __syncthreads();
  }
}

// =====================================================================
// Kernel 3: temporal attention, one block per batch. Projections +
// per-16-row score tiles with fp32 softmax; 288 WMMAs per block.
// =====================================================================
__global__ void __launch_bounds__(256) tempatt_kernel(
    const unsigned short* hf_g, const unsigned short* hb_g,
    const unsigned short* wt, const float* tvb, const float* tkb,
    const float* tqb, unsigned short* to_g) {
  __shared__ __align__(16) unsigned short sLo[NS * 64];
  __shared__ __align__(16) unsigned short sTv[NS * 64];
  __shared__ __align__(16) unsigned short sTkT[64 * NS];   // transposed K
  __shared__ __align__(16) unsigned short sTq[NS * 64];
  __shared__ float sS[16 * NS];
  __shared__ __align__(16) unsigned short sAtt[16 * NS];
  int b = blockIdx.x, t = threadIdx.x, wv = t >> 5, lane = t & 31;
  int m = lane & 15, h8 = (lane & 16) >> 1;
  const size_t base = (size_t)b * NS * 64;
  for (int i = t; i < NS * 64; i += 256)
    sLo[i] = f2bf(0.5f * (bf2f(hf_g[base + i]) + bf2f(hb_g[base + i])));
  __syncthreads();
  const float* tb[3] = {tvb, tkb, tqb};
  for (int job = wv; job < 72; job += 8) {       // 3 proj x 6 mt x 4 nt
    int p = job / 24, rem = job % 24, mt = rem >> 2, nt = rem & 3;
    v8f acc = {};
#pragma unroll
    for (int kc = 0; kc < 2; ++kc) {
      v16bf A  = load_A_lds(sLo + mt * 16 * 64 + kc * 32, 64);
      v16bf Bf = load_B_swz(wt + p * 4096 + (nt * 2 + kc) * 512);
      acc = WMMA_BF16(A, Bf, acc);
    }
    const float* bias = tb[p];
#pragma unroll
    for (int e = 0; e < 8; ++e) {
      int row = mt * 16 + e + h8, col = nt * 16 + m;
      unsigned short v = f2bf(acc[e] + bias[col]);
      if (p == 0) sTv[row * 64 + col] = v;
      else if (p == 1) sTkT[col * NS + row] = v;     // store tk transposed
      else sTq[row * 64 + col] = v;
    }
  }
  __syncthreads();
  for (int mt = 0; mt < 6; ++mt) {
    if (wv < 6) {                                  // scores tile [16 x 96]
      int nt = wv;
      v8f acc = {};
#pragma unroll
      for (int kc = 0; kc < 2; ++kc) {
        v16bf A  = load_A_lds(sTq + mt * 16 * 64 + kc * 32, 64);
        v16bf Bf = load_B_nat(sTkT + kc * 32 * NS + nt * 16, NS);
        acc = WMMA_BF16(A, Bf, acc);
      }
#pragma unroll
      for (int e = 0; e < 8; ++e) sS[(e + h8) * NS + nt * 16 + m] = acc[e];
    }
    __syncthreads();
    if (t < 16) {                                  // fp32 softmax over S
      float mx = -3.0e38f;
      for (int j = 0; j < NS; ++j) mx = fmaxf(mx, sS[t * NS + j]);
      float sum = 0.f;
      for (int j = 0; j < NS; ++j) sum += __expf(sS[t * NS + j] - mx);
      float inv = 1.0f / sum;
      for (int j = 0; j < NS; ++j) sAtt[t * NS + j] = f2bf(__expf(sS[t * NS + j] - mx) * inv);
    }
    __syncthreads();
    if (wv < 4) {                                  // to = tatt @ tv, tanh
      int nt = wv;
      v8f acc = {};
#pragma unroll
      for (int kc = 0; kc < 3; ++kc) {
        v16bf A  = load_A_lds(sAtt + kc * 32, NS);
        v16bf Bf = load_B_nat(sTv + kc * 32 * 64 + nt * 16, 64);
        acc = WMMA_BF16(A, Bf, acc);
      }
#pragma unroll
      for (int e = 0; e < 8; ++e) {
        int row = mt * 16 + e + h8;
        to_g[base + (size_t)row * 64 + nt * 16 + m] = f2bf(tanhf(acc[e]));
      }
    }
    __syncthreads();
  }
}

// =====================================================================
// Kernel 4: LSTM2 (64->16) + FC(16->7). grid NB/16, 128 thr (4 waves).
// Per-step input tile staged by the Tensor Data Mover: one 2D descriptor,
// 16 rows x 128B, LDS pad 64B/row to land on the 192B A-tile stride.
// =====================================================================
__global__ void __launch_bounds__(128) lstm2_kernel(
    const unsigned short* to_g, const unsigned short* wc2, const float* b2,
    const float* fcw, const float* fcb, float* out) {
  __shared__ __align__(16) unsigned short sA[16 * 96];
  __shared__ float sG[16 * 64];
  __shared__ float sC[16 * 16];
  __shared__ float sH[16 * 16];
  __shared__ float sFW[7 * 16];
  __shared__ float sFB[7];
  __shared__ float sB2[64];
  int t = threadIdx.x, wv = t >> 5, lane = t & 31;
  int m = lane & 15, h8 = (lane & 16) >> 1;
  int b0 = blockIdx.x * 16;
  __builtin_prefetch((const char*)wc2 + t * 96, 0, 3);   // warm 12KB weight blob
  for (int i = t; i < 16 * 96; i += 128) sA[i] = 0;
  for (int i = t; i < 256; i += 128) sC[i] = 0.f;
  if (t < 112) sFW[t] = fcw[t];
  if (t < 7)  sFB[t] = fcb[t];
  if (t < 64) sB2[t] = b2[t];
  // flat address low 32 bits == wave-relative LDS byte offset (ISA 10.2 aperture)
  unsigned lds_off = (unsigned)(unsigned long long)(const void*)&sA[0];
  __syncthreads();
  for (int s = 0; s < NS; ++s) {
    if (wv == 0) {   // TDM: stage to[16 x 64] bf16 into A-tile cols 0..63
      unsigned long long gaddr =
          (unsigned long long)(const void*)(to_g + ((size_t)b0 * NS + s) * 64);
      v4ui g0;
      g0[0] = 1u;                                   // count=1 (valid user D#)
      g0[1] = lds_off;                              // lds_addr
      g0[2] = (unsigned)(gaddr & 0xFFFFFFFFu);      // global_addr[31:0]
      g0[3] = (unsigned)((gaddr >> 32) & 0x01FFFFFFu) | (2u << 30);  // [56:32] | type=2
      v8i g1;
      g1[0] = (int)((1u << 16)      // data_size = 2B
                  | (1u << 20)      // pad_enable
                  | (4u << 22)      // pad_interval: 32 DWORDs (128B) per row
                  | (15u << 25));   // pad_amount: 16 DWORDs (64B) -> 192B stride
      g1[1] = (int)(64u << 16);                 // tensor_dim0 = 64 (bits 79:48)
      g1[2] = 0;                                // tensor_dim1 lo16 = 0 (NB*NS = 0x30000)
      g1[3] = (int)(3u | (64u << 16));          // tensor_dim1 hi | tile_dim0 = 64
      g1[4] = 16;                               // tile_dim1 = 16, tile_dim2 = 0
      g1[5] = NS * 64;                          // tensor_dim0_stride = 6144 elems
      g1[6] = 0; g1[7] = 0;
      v4i gz = {0, 0, 0, 0};
#if defined(__clang_major__) && (__clang_major__ >= 23)
      v8i gz8 = {0, 0, 0, 0, 0, 0, 0, 0};
      __builtin_amdgcn_tensor_load_to_lds(g0, g1, gz, gz, gz8, 0);
#else
      __builtin_amdgcn_tensor_load_to_lds(g0, g1, gz, gz, 0);
#endif
      __builtin_amdgcn_s_wait_tensorcnt(0);
    }
    __syncthreads();
    int nt = wv;
    v8f acc = {};
#pragma unroll
    for (int kc = 0; kc < 3; ++kc) {
      v16bf A  = load_A_lds(sA + kc * 32, 96);
      v16bf Bf = load_B_swz(wc2 + (nt * 3 + kc) * 512);
      acc = WMMA_BF16(A, Bf, acc);
    }
#pragma unroll
    for (int e = 0; e < 8; ++e) sG[(e + h8) * 64 + nt * 16 + m] = acc[e];
    __syncthreads();
    for (int u = t; u < 256; u += 128) {
      int r = u >> 4, j = u & 15;
      float gi = sG[r * 64 + j]      + sB2[j];
      float gf = sG[r * 64 + 16 + j] + sB2[16 + j];
      float gg = sG[r * 64 + 32 + j] + sB2[32 + j];
      float go = sG[r * 64 + 48 + j] + sB2[48 + j];
      float c = sigmoidf(gf) * sC[u] + sigmoidf(gi) * tanhf(gg);
      sC[u] = c;
      float h = sigmoidf(go) * tanhf(c);
      sH[u] = h;
      sA[r * 96 + 64 + j] = f2bf(h);
    }
    __syncthreads();
    if (t < 112) {   // FC 16 -> 7
      int r = t / 7, d = t % 7;
      float sum = sFB[d];
#pragma unroll
      for (int j = 0; j < 16; ++j) sum += sH[r * 16 + j] * sFW[d * 16 + j];
      out[((size_t)(b0 + r) * NS + s) * 7 + d] = sum;
    }
    __syncthreads();
  }
}

// =====================================================================
extern "C" void kernel_launch(void* const* d_in, const int* in_sizes, int n_in,
                              void* d_out, int out_size, void* d_ws, size_t ws_size,
                              hipStream_t stream) {
  (void)in_sizes; (void)n_in; (void)out_size; (void)ws_size;
  const float* x   = (const float*)d_in[0];
  const float* fvw = (const float*)d_in[1];  const float* fvb = (const float*)d_in[2];
  const float* fkw = (const float*)d_in[3];  const float* fkb = (const float*)d_in[4];
  const float* fqw = (const float*)d_in[5];  const float* fqb = (const float*)d_in[6];
  const float* l1f_wih = (const float*)d_in[7];  const float* l1f_whh = (const float*)d_in[8];
  const float* l1f_bih = (const float*)d_in[9];  const float* l1f_bhh = (const float*)d_in[10];
  const float* l1b_wih = (const float*)d_in[11]; const float* l1b_whh = (const float*)d_in[12];
  const float* l1b_bih = (const float*)d_in[13]; const float* l1b_bhh = (const float*)d_in[14];
  const float* tv_w = (const float*)d_in[15]; const float* tv_b = (const float*)d_in[16];
  const float* tk_w = (const float*)d_in[17]; const float* tk_b = (const float*)d_in[18];
  const float* tq_w = (const float*)d_in[19]; const float* tq_b = (const float*)d_in[20];
  const float* l2_wih = (const float*)d_in[21]; const float* l2_whh = (const float*)d_in[22];
  const float* l2_bih = (const float*)d_in[23]; const float* l2_bhh = (const float*)d_in[24];
  const float* fc_w = (const float*)d_in[25]; const float* fc_b = (const float*)d_in[26];
  float* out = (float*)d_out;

  unsigned char* p = (unsigned char*)d_ws;
  auto carve = [&](size_t bytes) -> void* {
    void* r = (void*)p;
    p += (bytes + 255) & ~((size_t)255);
    return r;
  };
  unsigned short* fa_g = (unsigned short*)carve((size_t)NB * NS * 8 * 2);
  unsigned short* hf_g = (unsigned short*)carve((size_t)NB * NS * 64 * 2);
  unsigned short* hb_g = (unsigned short*)carve((size_t)NB * NS * 64 * 2);
  unsigned short* to_g = (unsigned short*)carve((size_t)NB * NS * 64 * 2);
  unsigned short* wc1  = (unsigned short*)carve((size_t)2 * 24576 * 2);
  float*          b1   = (float*)carve(512 * 4);
  unsigned short* wc2  = (unsigned short*)carve((size_t)6144 * 2);
  float*          b2   = (float*)carve(64 * 4);
  unsigned short* wt   = (unsigned short*)carve((size_t)3 * 4096 * 2);

  prep_kernel<<<64, 256, 0, stream>>>(l1f_wih, l1f_whh, l1f_bih, l1f_bhh,
                                      l1b_wih, l1b_whh, l1b_bih, l1b_bhh,
                                      tv_w, tk_w, tq_w,
                                      l2_wih, l2_whh, l2_bih, l2_bhh,
                                      wc1, b1, wc2, b2, wt);
  featatt_kernel<<<NB, 128, 0, stream>>>(x, fvw, fvb, fkw, fkb, fqw, fqb, fa_g);
  lstm1_kernel<<<dim3(NB / 16, 2), 256, 0, stream>>>(fa_g, wc1, b1, hf_g, hb_g);
  tempatt_kernel<<<NB, 256, 0, stream>>>(hf_g, hb_g, wt, tv_b, tk_b, tq_b, to_g);
  lstm2_kernel<<<NB / 16, 128, 0, stream>>>(to_g, wc2, b2, fc_w, fc_b, out);
}
